// PostProcessor_19086834663978
// MI455X (gfx1250) — compile-verified, matrix operand order checked by reading, not run on hardware
//
#include <hip/hip_runtime.h>
#include <math.h>
#include <stdint.h>

// ---------------- constants (match reference) ----------------
#define N_PROP   1000
#define N_CLS    81
#define N_FG     80           // classes 1..80
#define N_TOT    (N_FG * N_PROP)   // 80000
#define NPAD     1024
#define TILE     256
#define SCORE_THRESH 0.05f
#define NMS_THRESH   0.5f
#define DET_PER_IMG  100
#define BBOX_CLIP    4.135166556742356f   // log(1000/16)
#define NEG_BIG      (-3.402823466e38f)

typedef _Float16 v16h __attribute__((ext_vector_type(16)));
typedef float    v8f  __attribute__((ext_vector_type(8)));

// ---------------- kernel 1: softmax -> class-major scores ----------------
__global__ __launch_bounds__(256) void softmax_kernel(const float* __restrict__ logits,
                                                      float* __restrict__ scoresw,
                                                      int* __restrict__ count) {
    int n = blockIdx.x * blockDim.x + threadIdx.x;
    if (blockIdx.x == 0 && threadIdx.x == 0) *count = 0;   // reset per launch
    if (n >= N_PROP) return;
    const float* row = logits + (size_t)n * N_CLS;
    float mx = row[0];
    for (int c = 1; c < N_CLS; ++c) mx = fmaxf(mx, row[c]);
    float sum = 0.0f;
    for (int c = 0; c < N_CLS; ++c) sum += expf(row[c] - mx);
    float inv = 1.0f / sum;
    for (int c = 1; c < N_CLS; ++c)
        scoresw[(size_t)(c - 1) * N_PROP + n] = expf(row[c] - mx) * inv;
}

// ---------------- kernel 2: decode + clip boxes ----------------
__global__ __launch_bounds__(256) void decode_kernel(const float* __restrict__ boxreg,
                                                     const float* __restrict__ props,
                                                     const int* __restrict__ imh,
                                                     const int* __restrict__ imw,
                                                     float* __restrict__ boxesw,
                                                     int* __restrict__ keepw) {
    int m = blockIdx.x * blockDim.x + threadIdx.x;
    if (m >= N_TOT) return;
    int c = m / N_PROP;          // 0..79  -> class c+1
    int n = m % N_PROP;
    float Wm1 = (float)imw[0] - 1.0f;
    float Hm1 = (float)imh[0] - 1.0f;

    const float* p = props + (size_t)n * 4;
    float w  = p[2] - p[0] + 1.0f;
    float h  = p[3] - p[1] + 1.0f;
    float cx = p[0] + 0.5f * w;
    float cy = p[1] + 0.5f * h;

    const float* r = boxreg + (size_t)n * (N_CLS * 4) + (size_t)(c + 1) * 4;
    float dx = r[0] / 10.0f;
    float dy = r[1] / 10.0f;
    float dw = fminf(r[2] / 5.0f, BBOX_CLIP);
    float dh = fminf(r[3] / 5.0f, BBOX_CLIP);

    float pcx = dx * w + cx;
    float pcy = dy * h + cy;
    float pw  = expf(dw) * w;
    float ph  = expf(dh) * h;

    float x1 = pcx - 0.5f * pw;
    float y1 = pcy - 0.5f * ph;
    float x2 = pcx + 0.5f * pw - 1.0f;
    float y2 = pcy + 0.5f * ph - 1.0f;

    x1 = fminf(fmaxf(x1, 0.0f), Wm1);
    y1 = fminf(fmaxf(y1, 0.0f), Hm1);
    x2 = fminf(fmaxf(x2, 0.0f), Wm1);
    y2 = fminf(fmaxf(y2, 0.0f), Hm1);

    float* o = boxesw + (size_t)m * 4;
    o[0] = x1; o[1] = y1; o[2] = x2; o[3] = y2;
    keepw[m] = 0;
}

// ---------------- kernel 3: per-class NMS (one WG per class) ----------------
__global__ __launch_bounds__(512) void nms_kernel(const float* __restrict__ scoresw,
                                                  const float* __restrict__ boxesw,
                                                  int* __restrict__ keepw,
                                                  int* __restrict__ count) {
    const int c   = blockIdx.x;
    const int tid = threadIdx.x;

    __shared__ float    skey[NPAD];
    __shared__ int      sidx[NPAD];
    __shared__ __attribute__((aligned(16))) float sbox[NPAD][4];
    __shared__ float    sarea[NPAD];
    __shared__ unsigned smask[TILE * 32];
    __shared__ unsigned kw[32];

    __builtin_prefetch(scoresw + (size_t)c * N_PROP, 0, 1);   // global_prefetch_b8

    // load keys: where(valid, score, -inf); pads get -inf and idx >= 1000
    for (int i = tid; i < NPAD; i += 512) {
        float key = NEG_BIG;
        if (i < N_PROP) {
            float s = scoresw[(size_t)c * N_PROP + i];
            key = (s > SCORE_THRESH) ? s : NEG_BIG;
        }
        skey[i] = key;
        sidx[i] = i;
    }
    __syncthreads();

    // bitonic sort, descending by key, ties broken by ascending idx (stable argsort)
    for (int k = 2; k <= NPAD; k <<= 1) {
        for (int j = k >> 1; j > 0; j >>= 1) {
            int i1 = ((tid & ~(j - 1)) << 1) | (tid & (j - 1));
            int i2 = i1 | j;
            float ka = skey[i1], kb = skey[i2];
            int   ia = sidx[i1], ib = sidx[i2];
            bool aAfter = (ka < kb) || (ka == kb && ia > ib); // a ranks after b (desc)
            bool descSeg = ((i1 & k) == 0);
            if (descSeg ? aAfter : !aAfter) {
                skey[i1] = kb; skey[i2] = ka;
                sidx[i1] = ib; sidx[i2] = ia;
            }
            __syncthreads();
        }
    }

    // gather sorted boxes into LDS via CDNA5 async-load-to-LDS (ASYNCcnt path)
    for (int i = tid; i < NPAD; i += 512) {
        int j = sidx[i]; if (j > N_PROP - 1) j = N_PROP - 1;
        unsigned long long ga  = (unsigned long long)(boxesw + ((size_t)c * N_PROP + j) * 4);
        unsigned           lda = (unsigned)(size_t)(&sbox[i][0]);
        asm volatile("global_load_async_to_lds_b128 %0, %1, off"
                     :: "v"(lda), "v"(ga) : "memory");
    }
    asm volatile("s_wait_asynccnt 0x0" ::: "memory");
    __syncthreads();

    for (int i = tid; i < NPAD; i += 512)
        sarea[i] = (sbox[i][2] - sbox[i][0] + 1.0f) * (sbox[i][3] - sbox[i][1] + 1.0f);

    // initial keep words from sorted validity
    if (tid < 32) {
        unsigned w = 0u;
        for (int b = 0; b < 32; ++b)
            if (skey[tid * 32 + b] > SCORE_THRESH) w |= (1u << b);
        kw[tid] = w;
    }
    __syncthreads();

    unsigned keep_reg = (tid < 32) ? kw[tid] : 0u;

    // tiled suppression matrix + sequential wave32 scan
    for (int t = 0; t < NPAD / TILE; ++t) {
        for (int widx = tid; widx < TILE * 32; widx += 512) {
            int li = widx >> 5;
            int w  = widx & 31;
            int gi = t * TILE + li;
            unsigned m = 0u;
            if (gi < N_PROP) {
                float x1 = sbox[gi][0], y1 = sbox[gi][1];
                float x2 = sbox[gi][2], y2 = sbox[gi][3];
                float ai = sarea[gi];
                int j0 = w * 32;
                if (j0 + 31 > gi) {
                    for (int b = 0; b < 32; ++b) {
                        int j = j0 + b;
                        if (j > gi) {
                            float iw = fminf(x2, sbox[j][2]) - fmaxf(x1, sbox[j][0]) + 1.0f;
                            float ih = fminf(y2, sbox[j][3]) - fmaxf(y1, sbox[j][1]) + 1.0f;
                            iw = fmaxf(iw, 0.0f);
                            ih = fmaxf(ih, 0.0f);
                            float inter = iw * ih;
                            float iou = inter / (ai + sarea[j] - inter);
                            if (iou > NMS_THRESH) m |= (1u << b);
                        }
                    }
                }
            }
            smask[widx] = m;
        }
        __syncthreads();
        if (tid < 32) {
            for (int li = 0; li < TILE; ++li) {
                int gi = t * TILE + li;
                if (gi >= N_PROP) break;
                unsigned wsrc = (unsigned)__shfl((int)keep_reg, gi >> 5);
                if ((wsrc >> (gi & 31)) & 1u)
                    keep_reg &= ~smask[li * 32 + tid];
            }
        }
        __syncthreads();
    }

    if (tid < 32) kw[tid] = keep_reg;
    __syncthreads();

    // scatter back to original indices + count kept
    int cnt = 0;
    for (int i = tid; i < N_PROP; i += 512) {
        int bit = (int)((kw[i >> 5] >> (i & 31)) & 1u);
        keepw[(size_t)c * N_PROP + sidx[i]] = bit;
        cnt += bit;
    }
    if (cnt) atomicAdd(count, cnt);
}

// ---------------- kernel 4: exact 101st-largest threshold (WMMA reduction) ---
__global__ __launch_bounds__(1024) void thresh_kernel(const float* __restrict__ scoresw,
                                                      const int* __restrict__ keepw,
                                                      const int* __restrict__ count,
                                                      float* __restrict__ threshf) {
    __shared__ float pts[1024];
    __shared__ float sh_cnt;
    int tid = threadIdx.x;

    unsigned lo = 0u, hi = 0x3F800000u;   // [0.0, 1.0] in ordered float-bit space
    while (lo < hi) {
        unsigned mid = lo + ((hi - lo + 1u) >> 1);
        float f = __uint_as_float(mid);
        int cnt = 0;
        for (int m = tid; m < N_TOT; m += 1024)
            if (keepw[m] && scoresw[m] >= f) ++cnt;
        pts[tid] = (float)cnt;              // <= 79, exact in f16
        __syncthreads();
        if (tid < 32) {
            v16h ones, b0, b1;
            for (int q = 0; q < 16; ++q) ones[q] = (_Float16)1.0f;
            for (int q = 0; q < 16; ++q) b0[q] = (_Float16)pts[tid * 16 + q];
            for (int q = 0; q < 16; ++q) b1[q] = (_Float16)pts[512 + tid * 16 + q];
            v8f acc = {};
            // D = ones(16x32) x B(32x16) + C  -> every column of D row0 is a colsum
            acc = __builtin_amdgcn_wmma_f32_16x16x32_f16(false, ones, false, b0,
                                                         (short)0, acc, false, false);
            acc = __builtin_amdgcn_wmma_f32_16x16x32_f16(false, ones, false, b1,
                                                         (short)0, acc, false, false);
            float fsum = acc[0];            // lanes 0..15 hold D[0][0..15]
            for (int off = 8; off >= 1; off >>= 1)
                fsum += __shfl_xor(fsum, off);
            if (tid == 0) sh_cnt = fsum;    // total count, exact
        }
        __syncthreads();
        int total = (int)sh_cnt;
        if (total >= DET_PER_IMG + 1) lo = mid; else hi = mid - 1;
        __syncthreads();
    }
    if (tid == 0) {
        int tc = *count;
        // topv[100] when count > 100, else -inf (use 0.0: kept scores > 0.05)
        threshf[0] = (tc > DET_PER_IMG) ? __uint_as_float(lo) : 0.0f;
    }
}

// ---------------- kernel 5: assemble outputs ----------------
__global__ __launch_bounds__(256) void output_kernel(const float* __restrict__ scoresw,
                                                     const float* __restrict__ boxesw,
                                                     const int* __restrict__ keepw,
                                                     const float* __restrict__ threshf,
                                                     float* __restrict__ out) {
    int m = blockIdx.x * blockDim.x + threadIdx.x;
    if (m >= N_TOT) return;
    int c = m / N_PROP;
    float s  = scoresw[m];
    float th = threshf[0];
    bool fin = (keepw[m] != 0) && (s >= th);

    float* o5 = out + (size_t)m * 5;
    const float* b = boxesw + (size_t)m * 4;
    o5[0] = fin ? b[0] : 0.0f;
    o5[1] = fin ? b[1] : 0.0f;
    o5[2] = fin ? b[2] : 0.0f;
    o5[3] = fin ? b[3] : 0.0f;
    o5[4] = fin ? s    : 0.0f;
    out[(size_t)N_TOT * 5 + m]         = fin ? (float)(c + 1) : 0.0f;  // labels
    out[(size_t)N_TOT * 5 + N_TOT + m] = fin ? 1.0f : 0.0f;            // final mask
}

// ---------------- host-side launcher ----------------
extern "C" void kernel_launch(void* const* d_in, const int* in_sizes, int n_in,
                              void* d_out, int out_size, void* d_ws, size_t ws_size,
                              hipStream_t stream) {
    const float* logits = (const float*)d_in[0];   // 1000 x 81
    const float* boxreg = (const float*)d_in[1];   // 1000 x 324
    const float* props  = (const float*)d_in[2];   // 1000 x 4
    const int*   imh    = (const int*)d_in[3];
    const int*   imw    = (const int*)d_in[4];
    float*       out    = (float*)d_out;

    float* ws      = (float*)d_ws;
    float* scoresw = ws;                       // 80000 floats  [class-major]
    float* boxesw  = ws + N_TOT;               // 320000 floats
    int*   keepw   = (int*)(ws + N_TOT * 5);   // 80000 ints
    int*   count   = (int*)(ws + N_TOT * 6);   // 1 int
    float* threshf = ws + N_TOT * 6 + 1;       // 1 float
    (void)in_sizes; (void)n_in; (void)out_size; (void)ws_size;

    softmax_kernel<<<(N_PROP + 255) / 256, 256, 0, stream>>>(logits, scoresw, count);
    decode_kernel<<<(N_TOT + 255) / 256, 256, 0, stream>>>(boxreg, props, imh, imw,
                                                           boxesw, keepw);
    nms_kernel<<<N_FG, 512, 0, stream>>>(scoresw, boxesw, keepw, count);
    thresh_kernel<<<1, 1024, 0, stream>>>(scoresw, keepw, count, threshf);
    output_kernel<<<(N_TOT + 255) / 256, 256, 0, stream>>>(scoresw, boxesw, keepw,
                                                           threshf, out);
}
